// CompositionalMapper_18691697672521
// MI455X (gfx1250) — 8-bit matrix operand order repaired and checked by reading, compiled, not run on hardware
//
#include <hip/hip_runtime.h>
#include <stdint.h>

#define N_BITS    64
#define N_GROUPS  8
#define P_BITS    8
#define MAX_TABLE 32768
#define BATCH     524288

typedef int   v8i __attribute__((ext_vector_type(8)));
typedef int   v4i __attribute__((ext_vector_type(4)));
typedef float v4f __attribute__((ext_vector_type(4)));

__device__ __forceinline__ uint32_t pack4(v4i v) {
  return (uint32_t)v.x | ((uint32_t)v.y << 8) | ((uint32_t)v.z << 16) | ((uint32_t)v.w << 24);
}

__global__ __launch_bounds__(256)
void comp_mapper_kernel(const int* __restrict__ bits,
                        const float* __restrict__ group_tables,
                        const int* __restrict__ carry_tables,
                        float* __restrict__ out)
{
  __shared__ int carr[(N_GROUPS - 1) * 256];
  __shared__ int glds[8][32][N_GROUPS];

  const int tid  = threadIdx.x;
  const int lane = tid & 31;
  const int wave = tid >> 5;

  for (int i = tid; i < 448; i += 256) {
    uint32_t ldsb = (uint32_t)(uintptr_t)(&carr[i * 4]);
    uint32_t goff = (uint32_t)(i * 16);
    asm volatile("global_load_async_to_lds_b128 %0, %1, %2"
                 :: "v"(ldsb), "v"(goff), "s"(carry_tables) : "memory");
  }
  asm volatile("s_wait_asynccnt 0" ::: "memory");
  __syncthreads();

  const int wavebase = blockIdx.x * 256 + wave * 32;
  const int L  = lane & 15;
  const int hi = lane >> 4;

  const v4i* bv = (const v4i*)bits;

  v8i a0, a1;
  #pragma unroll
  for (int t = 0; t < 8; ++t) {
    int chunk = ((t >> 1) << 2) + (t & 1) + (hi ? 2 : 0);
    v4i c0 = __builtin_nontemporal_load(&bv[(size_t)(wavebase + L) * 16 + chunk]);
    v4i c1 = __builtin_nontemporal_load(&bv[(size_t)(wavebase + 16 + L) * 16 + chunk]);
    a0[t] = (int)pack4(c0);
    a1[t] = (int)pack4(c1);
  }

  v8i b;
  #pragma unroll
  for (int j = 0; j < 8; ++j) {
    int grp = ((j >> 1) << 1) + (hi ? 1 : 0);
    b[j] = (L == grp) ? ((j & 1) ? 0x01020408 : 0x10204080) : 0;
  }

  v8i cz = {};
  v8i d0 = __builtin_amdgcn_wmma_i32_16x16x64_iu8(false, a0, false, b, cz, false, false);
  v8i d1 = __builtin_amdgcn_wmma_i32_16x16x64_iu8(false, a1, false, b, cz, false, false);

  if (L < N_GROUPS) {
    #pragma unroll
    for (int r = 0; r < 8; ++r) {
      glds[wave][r + hi * 8][L]      = d0[r];
      glds[wave][16 + r + hi * 8][L] = d1[r];
    }
  }
  __syncthreads();

  const int batch = wavebase + lane;
  int ga[N_GROUPS];
  #pragma unroll
  for (int g = 0; g < N_GROUPS; ++g) ga[g] = glds[wave][lane][g];

  float* outp = out + (size_t)batch * N_BITS;
  int cp = 0;
  #pragma unroll
  for (int g = 0; g < N_GROUPS; ++g) {
    int addr;
    if (g == 0) {
      addr = ga[0];
    } else {
      cp = (cp << 1) | carr[(g - 1) * 256 + ga[g - 1]];
      addr = (ga[g] << g) | cp;
    }
    const float* tbl = group_tables + (size_t)g * P_BITS * MAX_TABLE + addr;
    v4f v0, v1;
    v0.x = tbl[0 * MAX_TABLE]; v0.y = tbl[1 * MAX_TABLE];
    v0.z = tbl[2 * MAX_TABLE]; v0.w = tbl[3 * MAX_TABLE];
    v1.x = tbl[4 * MAX_TABLE]; v1.y = tbl[5 * MAX_TABLE];
    v1.z = tbl[6 * MAX_TABLE]; v1.w = tbl[7 * MAX_TABLE];
    __builtin_nontemporal_store(v0, (v4f*)(outp + g * 8));
    __builtin_nontemporal_store(v1, (v4f*)(outp + g * 8 + 4));
  }
}

extern "C" void kernel_launch(void* const* d_in, const int* in_sizes, int n_in,
                              void* d_out, int out_size, void* d_ws, size_t ws_size,
                              hipStream_t stream) {
  const int*   bits         = (const int*)d_in[0];
  const float* group_tables = (const float*)d_in[1];
  const int*   carry_tables = (const int*)d_in[2];
  float*       out          = (float*)d_out;

  dim3 grid(BATCH / 256), block(256);
  comp_mapper_kernel<<<grid, block, 0, stream>>>(bits, group_tables, carry_tables, out);
}
